// RNN_42838003810840
// MI455X (gfx1250) — matrix operand order repaired and checked by reading, not compiled, not run on hardware
//
#include <hip/hip_runtime.h>
#include <hip/hip_bf16.h>
#include <math.h>

typedef __attribute__((ext_vector_type(16))) _Float16 v16h;
typedef __attribute__((ext_vector_type(8)))  _Float16 v8h;
typedef __attribute__((ext_vector_type(8)))  float    v8f;
typedef __attribute__((ext_vector_type(4)))  float    f32x4;

static constexpr int Bb     = 64;
static constexpr int Ss     = 2048;
static constexpr int DIM    = 128;
static constexpr int LATENT = 256;
static constexpr int BS     = Bb * Ss;          // 131072 rows

// ---------------------------------------------------------------------------
// Kernel A: xp = x @ Wx^T + bx   written into d_out (to be overwritten by h).
// One wave = one 16-row M-tile; loops over 16 N-tiles; K=128 -> 4 WMMA k-chunks.
// Loads are staged into f32x4 arrays before conversion so the backend can
// batch the global_load_b128s instead of wait-per-load.
// ---------------------------------------------------------------------------
__global__ void __launch_bounds__(256) rnn_xp_kernel(
    const float* __restrict__ x, const float* __restrict__ Wx,
    const float* __restrict__ bx, float* __restrict__ out)
{
    const int lane  = threadIdx.x & 31;
    const int wave  = threadIdx.x >> 5;
    const int mtile = blockIdx.x * (blockDim.x >> 5) + wave;   // 0..8191 exact
    const int row0  = mtile * 16;
    const int g     = lane >> 4;      // K-group half
    const int hl    = lane & 15;      // A: row ; B/C: column

    // ---- A fragments: stage 16 b128 loads, then convert (CDNA5 A 16x32 layout)
    const float* xrow = x + (size_t)(row0 + hl) * DIM;
    f32x4 xa[16];
    #pragma unroll
    for (int kc = 0; kc < 4; ++kc) {
        const int kb = kc * 32 + g * 8;
        xa[kc * 4 + 0] = *(const f32x4*)(xrow + kb);
        xa[kc * 4 + 1] = *(const f32x4*)(xrow + kb + 4);
        xa[kc * 4 + 2] = *(const f32x4*)(xrow + kb + 16);
        xa[kc * 4 + 3] = *(const f32x4*)(xrow + kb + 20);
    }
    v16h a[4];
    #pragma unroll
    for (int kc = 0; kc < 4; ++kc)
        #pragma unroll
        for (int e = 0; e < 16; ++e)
            a[kc][e] = (_Float16)xa[kc * 4 + (e >> 2)][e & 3];

    #pragma unroll 1
    for (int nt = 0; nt < 16; ++nt) {
        const int n = nt * 16 + hl;
        const float* wrow = Wx + (size_t)n * DIM;   // B[k][n] = Wx[n][k] contiguous
        f32x4 w[16];
        #pragma unroll
        for (int kc = 0; kc < 4; ++kc)
            #pragma unroll
            for (int q = 0; q < 4; ++q)
                w[kc * 4 + q] = *(const f32x4*)(wrow + kc * 32 + g * 8 + 4 * (q & 1) + 16 * (q >> 1));
        v16h b[4];
        #pragma unroll
        for (int kc = 0; kc < 4; ++kc)
            #pragma unroll
            for (int e = 0; e < 16; ++e)
                b[kc][e] = (_Float16)w[kc * 4 + (e >> 2)][e & 3];

        v8f c = {};
        #pragma unroll
        for (int kc = 0; kc < 4; ++kc)
            c = __builtin_amdgcn_wmma_f32_16x16x32_f16(
                    false, a[kc], false, b[kc], (short)0, c, false, false);

        const float bias = bx[n];
        #pragma unroll
        for (int i = 0; i < 8; ++i) {
            const int m = i + 8 * g;                // C layout: lanes>=16 hold M=8..15
            out[(size_t)(row0 + m) * LATENT + n] = c[i] + bias;
        }
    }
}

// ---------------------------------------------------------------------------
// Kernel B: serial scan. 4 blocks x 512 threads; block b owns batch rows
// [16b, 16b+16). Wave w owns latent columns [16w, 16w+16). Wh fragments live
// in registers for all 2048 steps; h is double-buffered f16 in LDS.
// Per step: split barrier (s_barrier_signal -1 ... s_barrier_wait -1) hides
// the global h-store and the next step's xp prefetch behind barrier latency.
// ---------------------------------------------------------------------------
__global__ void __launch_bounds__(512, 1) rnn_scan_kernel(
    const float* __restrict__ h0, const float* __restrict__ Wh,
    const float* __restrict__ bh, float* __restrict__ out)
{
    const int lane  = threadIdx.x & 31;
    const int wave  = threadIdx.x >> 5;    // n-tile index 0..15
    const int g     = lane >> 4;
    const int hl    = lane & 15;
    const int bbase = blockIdx.x * 16;     // batch row base

    __shared__ __align__(16) _Float16 hbuf[2][16][LATENT];   // 16 KB

    // ---- Wh fragments (constant across steps): B[k][n] = Wh[n][k]
    const int n = wave * 16 + hl;
    v16h bw[8];
    {
        const float* wrow = Wh + (size_t)n * LATENT;
        #pragma unroll
        for (int kc = 0; kc < 8; ++kc) {
            f32x4 w0 = *(const f32x4*)(wrow + kc * 32 + g * 16);
            f32x4 w1 = *(const f32x4*)(wrow + kc * 32 + g * 16 + 4);
            f32x4 w2 = *(const f32x4*)(wrow + kc * 32 + g * 16 + 8);
            f32x4 w3 = *(const f32x4*)(wrow + kc * 32 + g * 16 + 12);
            #pragma unroll
            for (int j = 0; j < 4; ++j) {
                bw[kc][j]      = (_Float16)w0[j];
                bw[kc][4 + j]  = (_Float16)w1[j];
                bw[kc][8 + j]  = (_Float16)w2[j];
                bw[kc][12 + j] = (_Float16)w3[j];
            }
        }
    }
    const float bias = bh[n];

    // ---- h(0) -> LDS buffer 0
    for (int idx = threadIdx.x; idx < 16 * LATENT; idx += blockDim.x) {
        const int r = idx >> 8, k = idx & 255;
        hbuf[0][r][k] = (_Float16)h0[(size_t)(bbase + r) * LATENT + k];
    }
    __syncthreads();

    const size_t rstride = (size_t)Ss * LATENT;       // out stride per batch row
    // Base pointer for this lane's column at s=0, rows bbase+8g+0..7
    float* outp = out + ((size_t)(bbase + 8 * g) * Ss) * LATENT + n;

    // Prologue: prefetch xp for s=0
    float xpv[8];
    #pragma unroll
    for (int i = 0; i < 8; ++i) xpv[i] = outp[(size_t)i * rstride];

    int cur = 0;

    #pragma unroll 1
    for (int s = 0; s < Ss; ++s) {
        // ---- A fragments from LDS: issue all 16 b128 loads, one drain
        v16h a[8];
        #pragma unroll
        for (int kc = 0; kc < 8; ++kc) {
            const v8h* p = (const v8h*)&hbuf[cur][hl][kc * 32 + g * 16];
            v8h lo = p[0];      // K = kc*32 + g*16 + 0..7
            v8h hi = p[1];      // K = kc*32 + g*16 + 8..15
            #pragma unroll
            for (int j = 0; j < 8; ++j) { a[kc][j] = lo[j]; a[kc][8 + j] = hi[j]; }
        }

        // ---- two accumulator chains to halve WMMA RAW depth
        v8f c0 = {}, c1 = {};
        #pragma unroll
        for (int kc = 0; kc < 8; kc += 2) {
            c0 = __builtin_amdgcn_wmma_f32_16x16x32_f16(
                    false, a[kc],     false, bw[kc],     (short)0, c0, false, false);
            c1 = __builtin_amdgcn_wmma_f32_16x16x32_f16(
                    false, a[kc + 1], false, bw[kc + 1], (short)0, c1, false, false);
        }

        // ---- combine + tanh; write f16 h into the other LDS buffer
        float hv[8];
        #pragma unroll
        for (int i = 0; i < 8; ++i) {
            hv[i] = tanhf(c0[i] + c1[i] + xpv[i] + bias);
            hbuf[cur ^ 1][i + 8 * g][n] = (_Float16)hv[i];
        }

        // ---- split barrier: publish LDS writes, then overlap global traffic
        asm volatile("s_wait_dscnt 0x0" ::: "memory");
        asm volatile("s_barrier_signal -1" ::: "memory");

        // global f32 h store (nobody else reads these)
        #pragma unroll
        for (int i = 0; i < 8; ++i) outp[(size_t)i * rstride] = hv[i];

        // prefetch xp for next step (clamped: uniform control flow, EXEC stays 1s)
        float* outn = outp + ((s + 1 < Ss) ? LATENT : 0);
        #pragma unroll
        for (int i = 0; i < 8; ++i) xpv[i] = outn[(size_t)i * rstride];

        asm volatile("s_barrier_wait -1" ::: "memory");
        outp = outn;
        cur ^= 1;
    }
}

// ---------------------------------------------------------------------------
extern "C" void kernel_launch(void* const* d_in, const int* in_sizes, int n_in,
                              void* d_out, int out_size, void* d_ws, size_t ws_size,
                              hipStream_t stream)
{
    const float* x  = (const float*)d_in[0];   // [64,2048,128]
    const float* h0 = (const float*)d_in[1];   // [64,256]
    const float* Wx = (const float*)d_in[2];   // [256,128]
    const float* bx = (const float*)d_in[3];   // [256]
    const float* Wh = (const float*)d_in[4];   // [256,256]
    const float* bh = (const float*)d_in[5];   // [256]
    float* out = (float*)d_out;                // [64,2048,256]

    // Phase 1: xp into d_out. 8192 M-tiles / 8 waves per block = 1024 blocks.
    rnn_xp_kernel<<<dim3(BS / 16 / 8), dim3(256), 0, stream>>>(x, Wx, bx, out);

    // Phase 2: recurrent scan, 4 independent batch slices.
    rnn_scan_kernel<<<dim3(Bb / 16), dim3(512), 0, stream>>>(h0, Wh, bh, out);
}
